// WaveNet_69191923138824
// MI455X (gfx1250) — matrix operand order repaired and checked by reading, not compiled, not run on hardware
//
#include <hip/hip_runtime.h>
#include <hip/hip_bf16.h>
#include <stdint.h>

// ---------------- WMMA types ----------------
typedef __attribute__((ext_vector_type(16))) _Float16 v16h;
typedef __attribute__((ext_vector_type(8)))  float    v8f;

union V16H { v16h v; unsigned u[8]; };

#define NB 4
#define NT 16384
#define NC 64
#define NL 30

// ---- CDNA5 async copy global->LDS (ASYNCcnt path) ----
__device__ __forceinline__ void async_copy_b128(unsigned lds_off, const void* gptr) {
    unsigned long long ga = (unsigned long long)(uintptr_t)gptr;
    asm volatile("global_load_async_to_lds_b128 %0, %1, off"
                 :: "v"(lds_off), "v"(ga) : "memory");
}
__device__ __forceinline__ void wait_async0() {
#if __has_builtin(__builtin_amdgcn_s_wait_asynccnt)
    __builtin_amdgcn_s_wait_asynccnt(0);
#else
    asm volatile("s_wait_asynccnt 0x0" ::: "memory");
#endif
}

// Load A fragment (16x32 f16) from an f16 row-major [rows][rstride] LDS array.
// Layout: lanes0-15 M=lane, K=0..15 (2 per VGPR);
//         lanes16-31 same M, K=16..31.
__device__ __forceinline__ v16h load_a_frag(const _Float16* base, int m0, int kc, int rstride) {
    int lane = threadIdx.x & 31;
    int m  = m0 + (lane & 15);
    int kb = (lane >> 4) * 16;
    const _Float16* row = base + m * rstride + kc;
    V16H r;
#pragma unroll
    for (int j = 0; j < 8; ++j)
        r.u[j] = *(const unsigned*)(row + kb + 2 * j);
    return r.v;
}

// Load B fragment (32x16 f16) from transposed weights wT[cout][cin] (cin contiguous).
// lanes0-15: N=lane, K=0..15 (2 per VGPR); lanes16-31: same N, K=16..31.
__device__ __forceinline__ v16h load_b_frag(const _Float16* wT, int n0, int kc, int rstride) {
    int lane = threadIdx.x & 31;
    int n  = n0 + (lane & 15);
    int kh = (lane >> 4) * 16;
    const _Float16* row = wT + n * rstride + kc + kh;
    V16H r;
#pragma unroll
    for (int j = 0; j < 8; ++j)
        r.u[j] = *(const unsigned*)(row + 2 * j);
    return r.v;
}

// ---------------- prep kernels ----------------
__global__ __launch_bounds__(256) void x0_kernel(const float* __restrict__ signal,
                                                 const float* __restrict__ in_w,
                                                 const float* __restrict__ in_b,
                                                 float* __restrict__ x,
                                                 _Float16* __restrict__ xh) {
    long i = (long)blockIdx.x * 256 + threadIdx.x;       // over B*T*C
    if (i >= (long)NB * NT * NC) return;
    int c = (int)(i & 63);
    long bt = i >> 6;
    float v = signal[bt] * in_w[c] + in_b[c];
    v = v > 0.f ? v : 0.f;
    x[i]  = v;
    xh[i] = (_Float16)v;
}

__global__ __launch_bounds__(256) void wconvT_kernel(const float* __restrict__ w, _Float16* __restrict__ wT) {
    // w: [L][3][64][128] -> wT: [L][3][128][64]
    long i = (long)blockIdx.x * 256 + threadIdx.x;
    if (i >= (long)NL * 3 * 64 * 128) return;
    long lt = i / (64 * 128);
    int rem = (int)(i % (64 * 128));
    int cin = rem / 128, cout = rem % 128;
    wT[lt * 128 * 64 + cout * 64 + cin] = (_Float16)w[i];
}

__global__ __launch_bounds__(256) void wmatT_kernel(const float* __restrict__ w, _Float16* __restrict__ wT, int nmat) {
    // w: [nmat][64 cin][64 cout] -> wT: [nmat][cout][cin]
    long i = (long)blockIdx.x * 256 + threadIdx.x;
    if (i >= (long)nmat * 4096) return;
    long m = i >> 12;
    int rem = (int)(i & 4095);
    int cin = rem >> 6, cout = rem & 63;
    wT[m * 4096 + cout * 64 + cin] = (_Float16)w[i];
}

__global__ __launch_bounds__(512) void embed_kernel(const int* __restrict__ timestep,
                                                    const float* __restrict__ pe0_w, const float* __restrict__ pe0_b,
                                                    const float* __restrict__ pe1_w, const float* __restrict__ pe1_b,
                                                    float* __restrict__ embed /* [4][512] */) {
    __shared__ float e0[NB][128];
    __shared__ float h1[NB][512];
    int tid = threadIdx.x;
    {
        int b = tid >> 7, j = tid & 127;
        float t = (float)timestep[b];
        int jj = j & 63;
        float logit = (float)jj * (1.0f / 63.0f);
        float ex = __expf(logit * 4.0f * 2.302585093f);   // 10^(4*logit)
        float comp = ex * t;
        e0[b][j] = (j < 64) ? __sinf(comp) : __cosf(comp);
    }
    __syncthreads();
    for (int idx = tid; idx < NB * 512; idx += 512) {
        int b = idx >> 9, o = idx & 511;
        float s = pe0_b[o];
        for (int k = 0; k < 128; ++k) s += e0[b][k] * pe0_w[k * 512 + o];
        h1[b][o] = s / (1.f + __expf(-s));                // swish
    }
    __syncthreads();
    for (int idx = tid; idx < NB * 512; idx += 512) {
        int b = idx >> 9, o = idx & 511;
        float s = pe1_b[o];
        for (int k = 0; k < 512; ++k) s += h1[b][k] * pe1_w[k * 512 + o];
        embed[idx] = s / (1.f + __expf(-s));
    }
}

__global__ __launch_bounds__(256) void eproj_kernel(const float* __restrict__ embed,
                                                    const float* __restrict__ emb_w,  // [L][512][64]
                                                    const float* __restrict__ emb_b,  // [L][64]
                                                    float* __restrict__ eL /* [L][4][64] */) {
    int l = blockIdx.x;
    int tid = threadIdx.x;
    int b = tid >> 6, c = tid & 63;
    const float* w = emb_w + (long)l * 512 * 64;
    const float* e = embed + b * 512;
    float s = emb_b[l * 64 + c];
    for (int k = 0; k < 512; ++k) s += e[k] * w[k * 64 + c];
    eL[(l * NB + b) * 64 + c] = s;
}

// ebias[l][b][tap][128] = e[l][b] @ conv_w[l][tap]   (fp32, folds e out of conv input)
__global__ __launch_bounds__(512) void ebias_kernel(const float* __restrict__ eL,
                                                    const float* __restrict__ conv_w, // [L][3][64][128]
                                                    float* __restrict__ ebias) {
    int l = blockIdx.x;
    int tid = threadIdx.x;
    for (int idx = tid; idx < NB * 3 * 128; idx += 512) {
        int b = idx / 384;
        int rem = idx % 384;
        int tap = rem >> 7, cout = rem & 127;
        const float* e = eL + (l * NB + b) * 64;
        const float* w = conv_w + ((long)(l * 3 + tap) * 64) * 128 + cout;
        float s = 0.f;
        for (int c = 0; c < 64; ++c) s += e[c] * w[c * 128];
        ebias[(long)(l * NB + b) * 384 + rem] = s;
    }
}

// ---------------- fused per-layer kernel ----------------
__global__ __launch_bounds__(256) void layer_kernel(
    const float* __restrict__ x_in, float* __restrict__ x_out,
    const _Float16* __restrict__ xh_in, _Float16* __restrict__ xh_out,
    float* __restrict__ skip_sum,
    const float* __restrict__ ebias,     // [L][4][3][128]
    const _Float16* __restrict__ convT,  // [L][3][128][64]
    const float* __restrict__ conv_b,    // [L][128]
    const _Float16* __restrict__ resT,   // [L-1][64][64]
    const float* __restrict__ res_b,     // [L-1][64]
    const _Float16* __restrict__ skipT,  // [L][64][64]
    const float* __restrict__ skip_b,    // [L][64]
    int layer, int dil, int has_res) {

    __shared__ _Float16 hA[3][64][64];   // x strips per tap (dilation-shifted), zero-padded
    __shared__ float    sigT[64][64];    // sigmoid half exchange
    __shared__ _Float16 zA[64][64];      // gated output, f16 for stage-2 GEMMs

    const int tid = threadIdx.x;
    const int b  = blockIdx.y;
    const long t0 = (long)blockIdx.x * 64;

    // ---- stage x strips into LDS via async DMA (pure f16 copy; SAME padding => zeros)
    {
        unsigned lds_base = (unsigned)(uintptr_t)(&hA[0][0][0]);
#pragma unroll
        for (int i = 0; i < 6; ++i) {
            int idx  = tid + i * 256;           // 1536 chunks of 16B
            int tap  = idx >> 9;
            int rem  = idx & 511;
            int row  = rem >> 3;
            int part = rem & 7;                 // 8 x 16B per 128B row
            long t = t0 + row + (long)dil * (tap - 1);
            unsigned loff = lds_base + (unsigned)(idx * 16);
            if (t >= 0 && t < NT) {
                async_copy_b128(loff, xh_in + ((((long)b * NT + t) << 6) + part * 8));
            } else {
                int4 z = {0, 0, 0, 0};
                *(int4*)(&hA[tap][row][part * 8]) = z;    // ds_store_b128 of zeros
            }
        }
        wait_async0();
    }
    __syncthreads();

    const int wave  = tid >> 5;
    const int lane  = tid & 31;
    const int lm    = 8 * (lane >> 4);     // row offset of this lane's D registers
    const int ln    = lane & 15;           // column within tile
    const int mtile = wave & 3;            // 4 M tiles of 16 time steps
    const int nhalf = wave >> 2;           // 0: tanh half (cout 0-63), 1: sigmoid half

    // ---- stage 1: dilated conv, y = sum_tap x_tap @ W_tap  (64 -> 128)
    v8f acc[4] = {};
    const _Float16* wc = convT + (long)layer * 3 * 128 * 64;
#pragma unroll
    for (int tap = 0; tap < 3; ++tap) {
#pragma unroll
        for (int kc = 0; kc < 64; kc += 32) {
            v16h a = load_a_frag(&hA[tap][0][0], mtile * 16, kc, 64);
#pragma unroll
            for (int nt = 0; nt < 4; ++nt) {
                v16h bb = load_b_frag(wc + tap * 128 * 64, (nhalf * 4 + nt) * 16, kc, 64);
                acc[nt] = __builtin_amdgcn_wmma_f32_16x16x32_f16(
                    false, a, false, bb, (short)0, acc[nt], false, false);
            }
        }
    }

    // ---- epilogue bias: conv_b + e@W1 + (in-range ? e@W0) + (in-range ? e@W2)
    const float* eb = ebias + (long)(layer * NB + b) * 384;  // [3][128]

    // ---- gate: z = tanh(y[:,:64]) * sigmoid(y[:,64:])
    if (nhalf == 1) {   // wave-uniform
#pragma unroll
        for (int nt = 0; nt < 4; ++nt) {
            int cout = 64 + nt * 16 + ln;
            float b1 = conv_b[layer * 128 + cout] + eb[128 + cout];
            float b0 = eb[cout], b2 = eb[256 + cout];
#pragma unroll
            for (int r = 0; r < 8; ++r) {
                long t = t0 + mtile * 16 + lm + r;
                float yv = acc[nt][r] + b1 + (t >= dil ? b0 : 0.f) + (t + dil < NT ? b2 : 0.f);
                sigT[mtile * 16 + lm + r][nt * 16 + ln] = 1.f / (1.f + __expf(-yv));
            }
        }
    }
    __syncthreads();
    if (nhalf == 0) {   // wave-uniform
#pragma unroll
        for (int nt = 0; nt < 4; ++nt) {
            int cout = nt * 16 + ln;
            float b1 = conv_b[layer * 128 + cout] + eb[128 + cout];
            float b0 = eb[cout], b2 = eb[256 + cout];
#pragma unroll
            for (int r = 0; r < 8; ++r) {
                long t = t0 + mtile * 16 + lm + r;
                float yv = acc[nt][r] + b1 + (t >= dil ? b0 : 0.f) + (t + dil < NT ? b2 : 0.f);
                float z = tanhf(yv) * sigT[mtile * 16 + lm + r][cout];
                zA[mtile * 16 + lm + r][cout] = (_Float16)z;
            }
        }
    }
    __syncthreads();

    // ---- stage 2: res (waves 0-3) and skip (waves 4-7) GEMMs, z[64x64] @ W[64x64]
    const _Float16* w2 = (nhalf == 0) ? (resT + (long)layer * 4096) : (skipT + (long)layer * 4096);
    const float*    b2p = (nhalf == 0) ? (res_b + layer * 64)       : (skip_b + layer * 64);
    const bool active = (nhalf == 1) || (has_res != 0);   // wave-uniform

    if (active) {
        v8f acc2[4] = {};
#pragma unroll
        for (int kc = 0; kc < 64; kc += 32) {
            v16h a = load_a_frag(&zA[0][0], mtile * 16, kc, 64);
#pragma unroll
            for (int nt = 0; nt < 4; ++nt) {
                v16h bb = load_b_frag(w2, nt * 16, kc, 64);
                acc2[nt] = __builtin_amdgcn_wmma_f32_16x16x32_f16(
                    false, a, false, bb, (short)0, acc2[nt], false, false);
            }
        }
#pragma unroll
        for (int nt = 0; nt < 4; ++nt) {
            int c = nt * 16 + ln;
            float bias = b2p[c];
#pragma unroll
            for (int r = 0; r < 8; ++r) {
                int m = mtile * 16 + lm + r;
                long off = (((long)b * NT + t0 + m) << 6) + c;
                float v = acc2[nt][r] + bias;
                if (nhalf == 0) {
                    float xn = (v + x_in[off]) * 0.70710678118f;   // residual / sqrt(2)
                    x_out[off]  = xn;
                    xh_out[off] = (_Float16)xn;                    // f16 mirror for next conv
                } else {
                    skip_sum[off] += v;
                }
            }
        }
    }
}

// ---------------- output projection ----------------
__global__ __launch_bounds__(128) void out_kernel(const float* __restrict__ skip,
                                                  const _Float16* __restrict__ out0T,  // [64][64]
                                                  const float* __restrict__ out0_b,
                                                  const float* __restrict__ out1_w,    // [64]
                                                  const float* __restrict__ out1_b,
                                                  float* __restrict__ out) {
    __shared__ _Float16 cA[64][64];
    __shared__ float    rl[64][64];
    const int tid = threadIdx.x;
    const int b = blockIdx.y;
    const long t0 = (long)blockIdx.x * 64;
    const float scale = 0.18257418583f;  // 1/sqrt(30)

    for (int idx = tid; idx < 4096; idx += 128) {
        int m = idx >> 6, c = idx & 63;
        cA[m][c] = (_Float16)(skip[(((long)b * NT + t0 + m) << 6) + c] * scale);
    }
    __syncthreads();

    const int wave = tid >> 5;
    const int lane = tid & 31;
    const int lm = 8 * (lane >> 4), ln = lane & 15;

    v8f acc[4] = {};
#pragma unroll
    for (int kc = 0; kc < 64; kc += 32) {
        v16h a = load_a_frag(&cA[0][0], wave * 16, kc, 64);
#pragma unroll
        for (int nt = 0; nt < 4; ++nt) {
            v16h bb = load_b_frag(out0T, nt * 16, kc, 64);
            acc[nt] = __builtin_amdgcn_wmma_f32_16x16x32_f16(
                false, a, false, bb, (short)0, acc[nt], false, false);
        }
    }
#pragma unroll
    for (int nt = 0; nt < 4; ++nt) {
        int c = nt * 16 + ln;
        float bias = out0_b[c];
#pragma unroll
        for (int r = 0; r < 8; ++r) {
            float v = acc[nt][r] + bias;
            rl[wave * 16 + lm + r][c] = v > 0.f ? v : 0.f;
        }
    }
    __syncthreads();
    if (tid < 64) {
        float s = out1_b[0];
        for (int n = 0; n < 64; ++n) s += rl[tid][n] * out1_w[n];
        out[(long)b * NT + t0 + tid] = s;
    }
}

// ---------------- launch ----------------
extern "C" void kernel_launch(void* const* d_in, const int* in_sizes, int n_in,
                              void* d_out, int out_size, void* d_ws, size_t ws_size,
                              hipStream_t stream) {
    const float* signal     = (const float*)d_in[0];
    const int*   timestep   = (const int*)d_in[1];
    const float* in_w       = (const float*)d_in[2];
    const float* in_b       = (const float*)d_in[3];
    const float* pe0_w      = (const float*)d_in[4];
    const float* pe0_b      = (const float*)d_in[5];
    const float* pe1_w      = (const float*)d_in[6];
    const float* pe1_b      = (const float*)d_in[7];
    const float* blk_emb_w  = (const float*)d_in[8];
    const float* blk_emb_b  = (const float*)d_in[9];
    const float* blk_conv_w = (const float*)d_in[10];
    const float* blk_conv_b = (const float*)d_in[11];
    const float* blk_res_w  = (const float*)d_in[12];
    const float* blk_res_b  = (const float*)d_in[13];
    const float* blk_skip_w = (const float*)d_in[14];
    const float* blk_skip_b = (const float*)d_in[15];
    const float* out0_w     = (const float*)d_in[16];
    const float* out0_b     = (const float*)d_in[17];
    const float* out1_w     = (const float*)d_in[18];
    const float* out1_b     = (const float*)d_in[19];

    // workspace carve-out (~70 MB)
    size_t off = 0;
    char* base = (char*)d_ws;
    auto carve = [&](size_t bytes) -> char* {
        char* p = base + off;
        off += (bytes + 255) & ~(size_t)255;
        return p;
    };
    const size_t XB  = (size_t)NB * NT * NC * sizeof(float);
    const size_t XHB = (size_t)NB * NT * NC * sizeof(_Float16);
    float*    x_a   = (float*)carve(XB);
    float*    x_b   = (float*)carve(XB);
    float*    skip  = (float*)carve(XB);
    _Float16* xh_a  = (_Float16*)carve(XHB);
    _Float16* xh_b  = (_Float16*)carve(XHB);
    float*    embed = (float*)carve((size_t)NB * 512 * sizeof(float));
    float*    eL    = (float*)carve((size_t)NL * NB * 64 * sizeof(float));
    float*    ebias = (float*)carve((size_t)NL * NB * 3 * 128 * sizeof(float));
    _Float16* convT = (_Float16*)carve((size_t)NL * 3 * 128 * 64 * sizeof(_Float16));
    _Float16* resT  = (_Float16*)carve((size_t)(NL - 1) * 4096 * sizeof(_Float16));
    _Float16* skipT = (_Float16*)carve((size_t)NL * 4096 * sizeof(_Float16));
    _Float16* out0T = (_Float16*)carve((size_t)4096 * sizeof(_Float16));
    (void)ws_size; (void)in_sizes; (void)n_in; (void)out_size;

    // prep
    {
        long n = (long)NB * NT * NC;
        x0_kernel<<<dim3((unsigned)((n + 255) / 256)), 256, 0, stream>>>(signal, in_w, in_b, x_a, xh_a);
    }
    hipMemsetAsync(skip, 0, XB, stream);
    {
        long n = (long)NL * 3 * 64 * 128;
        wconvT_kernel<<<dim3((unsigned)((n + 255) / 256)), 256, 0, stream>>>(blk_conv_w, convT);
    }
    wmatT_kernel<<<dim3((unsigned)(((NL - 1) * 4096 + 255) / 256)), 256, 0, stream>>>(blk_res_w, resT, NL - 1);
    wmatT_kernel<<<dim3((unsigned)((NL * 4096 + 255) / 256)), 256, 0, stream>>>(blk_skip_w, skipT, NL);
    wmatT_kernel<<<dim3((unsigned)((4096 + 255) / 256)), 256, 0, stream>>>(out0_w, out0T, 1);
    embed_kernel<<<1, 512, 0, stream>>>(timestep, pe0_w, pe0_b, pe1_w, pe1_b, embed);
    eproj_kernel<<<NL, 256, 0, stream>>>(embed, blk_emb_w, blk_emb_b, eL);
    ebias_kernel<<<NL, 512, 0, stream>>>(eL, blk_conv_w, ebias);

    // 30 residual layers, ping-pong x / xh
    for (int i = 0; i < NL; ++i) {
        int dil = 1 << (i % 10);
        const float*    xin  = (i & 1) ? x_b : x_a;
        float*          xout = (i & 1) ? x_a : x_b;
        const _Float16* xhin  = (i & 1) ? xh_b : xh_a;
        _Float16*       xhout = (i & 1) ? xh_a : xh_b;
        layer_kernel<<<dim3(NT / 64, NB), 256, 0, stream>>>(
            xin, xout, xhin, xhout, skip, ebias, convT, blk_conv_b,
            resT, blk_res_b, skipT, blk_skip_b, i, dil, (i < NL - 1) ? 1 : 0);
    }

    // output projection
    out_kernel<<<dim3(NT / 64, NB), 128, 0, stream>>>(skip, out0T, out0_b, out1_w, out1_b, (float*)d_out);
}
